// Atom2BondLayer_1_41532333752513
// MI455X (gfx1250) — matrix operand order repaired and checked by reading, not compiled, not run on hardware
//
#include <hip/hip_runtime.h>

// ---------------------------------------------------------------------------
// Atom2Bond: out = relu(concat(atom[src_idx], edge) @ W + b)
// M=640000, N=128, K=192.  bf16x3-split WMMA GEMM (f32-class accuracy) on
// CDNA5 v_wmma_f32_16x16x32_bf16, gather+convert staged through LDS.
// ---------------------------------------------------------------------------

typedef __attribute__((ext_vector_type(16))) __bf16 v16bf;
typedef __attribute__((ext_vector_type(8)))  float  v8f;

#define KDIM 192
#define NDIM 128
#define MBLK 128
#define SA   200   // padded K-stride in ushorts (400 B, multiple of 16 B)

struct alignas(16) U128 { unsigned int x, y, z, w; };
struct alignas(16) F128 { float x, y, z, w; };
struct alignas(8)  U64  { unsigned int x, y; };

__device__ __forceinline__ unsigned short bf16_rn(float x) {
  unsigned int u = __float_as_uint(x);
  u += 0x7FFFu + ((u >> 16) & 1u);           // round-to-nearest-even
  return (unsigned short)(u >> 16);
}
__device__ __forceinline__ void bf16_split(float x, unsigned short& h, unsigned short& l) {
  h = bf16_rn(x);
  float hf = __uint_as_float(((unsigned int)h) << 16);
  l = bf16_rn(x - hf);
}

union V16U { U128 q[2]; v16bf v; };

// A fragment (16x32 bf16): lane (l&15)=row, lane>>4 selects K-halves.
// Per ISA: elements 0..7 = K+khalf..K+khalf+7, elements 8..15 = K+16+khalf..
__device__ __forceinline__ v16bf ldfragA(const unsigned short* p) {
  V16U u;
  u.q[0] = *(const U128*)p;          // 8 bf16 at K
  u.q[1] = *(const U128*)(p + 16);   // 8 bf16 at K+16
  return u.v;
}
// B fragment (32x16 bf16): lane (l&15)=column, same K order as A:
// elements 0..7 = K+khalf..K+khalf+7, elements 8..15 = K+16+khalf..
__device__ __forceinline__ v16bf ldfragB(const unsigned short* p) {
  V16U u;
  u.q[0] = *(const U128*)p;
  u.q[1] = *(const U128*)(p + 16);
  return u.v;
}

#define WMMA_BF16(A, B, C) \
  __builtin_amdgcn_wmma_f32_16x16x32_bf16(false, (A), false, (B), (short)0, (C), false, false)

// ---- one-time W split: W[192][128] f32 -> ws: Wt_hi[n][192], Wt_lo[n][192] bf16
__global__ void wprep_kernel(const float* __restrict__ W, unsigned short* __restrict__ ws) {
  for (int idx = threadIdx.x; idx < KDIM * NDIM; idx += 256) {
    int k = idx >> 7;            // row of W (input dim)
    int n = idx & (NDIM - 1);    // col of W (output dim)
    unsigned short h, l;
    bf16_split(W[idx], h, l);
    ws[n * KDIM + k]              = h;
    ws[KDIM * NDIM + n * KDIM + k] = l;
  }
}

__global__ __launch_bounds__(256, 1)
void a2b_kernel(const float* __restrict__ atom,
                const float* __restrict__ edge,
                const int*   __restrict__ src_idx,
                const float* __restrict__ W,
                const float* __restrict__ bias,
                const unsigned short* __restrict__ wsplit,
                float* __restrict__ out,
                int n_edges, int use_ws) {
  extern __shared__ unsigned short lds[];
  unsigned short* Ahi = lds;                 // [MBLK][SA]
  unsigned short* Alo = lds + 1 * MBLK * SA;
  unsigned short* Whi = lds + 2 * MBLK * SA; // [NDIM][SA] (transposed: [n][k])
  unsigned short* Wlo = lds + 3 * MBLK * SA;

  const int t  = threadIdx.x;
  const int m0 = blockIdx.x * MBLK;

  // ---- stage W (bf16 hi/lo, [n][k]) into LDS ----
  if (use_ws) {
#pragma unroll
    for (int i = 0; i < 12; ++i) {           // 3072 16-byte chunks per array
      int c = t + i * 256;
      int r = c / 24, j = c - r * 24;        // row n, 8-ushort chunk j
      U128 hv = ((const U128*)wsplit)[c];
      U128 lv = ((const U128*)wsplit)[c + 3072];
      *(U128*)&Whi[r * SA + j * 8] = hv;
      *(U128*)&Wlo[r * SA + j * 8] = lv;
    }
  } else {                                   // fallback: split f32 W per block
    for (int i = 0; i < 96; ++i) {
      int idx = t + i * 256;
      int k = idx >> 7, n = idx & (NDIM - 1);
      unsigned short h, l;
      bf16_split(W[idx], h, l);
      Whi[n * SA + k] = h;
      Wlo[n * SA + k] = l;
    }
  }

  // ---- gather + bf16-split A rows into LDS: 2 threads per row, 96 K each ----
  {
    int r    = t >> 1;
    int half = t & 1;
    int m    = m0 + r;
    bool valid = (m < n_edges);
    size_t src = valid ? (size_t)src_idx[m] : 0;
    const float* nodep = atom + src * NDIM;
    const float* edgep = edge + (size_t)m * 64;
    int kb = half * 96;
#pragma unroll
    for (int kk = 0; kk < 96; kk += 4) {
      int k = kb + kk;
      F128 v;
      if (!valid)      { v.x = v.y = v.z = v.w = 0.0f; }
      else if (k < NDIM) v = *(const F128*)(nodep + k);
      else               v = *(const F128*)(edgep + (k - NDIM));
      unsigned short h0,h1,h2,h3,l0,l1,l2,l3;
      bf16_split(v.x, h0, l0); bf16_split(v.y, h1, l1);
      bf16_split(v.z, h2, l2); bf16_split(v.w, h3, l3);
      U64 hp; hp.x = (unsigned)h0 | ((unsigned)h1 << 16); hp.y = (unsigned)h2 | ((unsigned)h3 << 16);
      U64 lp; lp.x = (unsigned)l0 | ((unsigned)l1 << 16); lp.y = (unsigned)l2 | ((unsigned)l3 << 16);
      *(U64*)&Ahi[r * SA + k] = hp;
      *(U64*)&Alo[r * SA + k] = lp;
    }
  }
  __syncthreads();

  // ---- wave tiling: 8 waves -> (wm 0..3) x (wn 0..1); wave tile 32x64 ----
  const int lane = t & 31;
  const int w    = t >> 5;
  const int wm   = w >> 1;
  const int wn   = w & 1;
  const int lr   = lane & 15;
  const int lh   = lane >> 4;

  const int arow0 = wm * 32 + lr;   // + mi*16
  const int khalf = lh * 8;         // A K-half select
  const int bn0   = wn * 64 + lr;   // + ni*16
  const int bk    = lh * 8;         // B K-half select

  v8f acc[2][4];
#pragma unroll
  for (int mi = 0; mi < 2; ++mi)
#pragma unroll
    for (int ni = 0; ni < 4; ++ni) acc[mi][ni] = {};

#pragma unroll
  for (int kb = 0; kb < KDIM; kb += 32) {
    v16bf ah[2], al[2];
#pragma unroll
    for (int mi = 0; mi < 2; ++mi) {
      int off = (arow0 + mi * 16) * SA + kb + khalf;
      ah[mi] = ldfragA(&Ahi[off]);
      al[mi] = ldfragA(&Alo[off]);
    }
#pragma unroll
    for (int ni = 0; ni < 4; ++ni) {
      int off = (bn0 + ni * 16) * SA + kb + bk;
      v16bf bh = ldfragB(&Whi[off]);
      v16bf bl = ldfragB(&Wlo[off]);
#pragma unroll
      for (int mi = 0; mi < 2; ++mi) {
        // hi*hi + lo*hi + hi*lo  ~= full f32 product, accumulated in f32
        acc[mi][ni] = WMMA_BF16(ah[mi], bh, acc[mi][ni]);
        acc[mi][ni] = WMMA_BF16(al[mi], bh, acc[mi][ni]);
        acc[mi][ni] = WMMA_BF16(ah[mi], bl, acc[mi][ni]);
      }
    }
  }

  // ---- epilogue: + bias, relu, store (C/D layout: vgpr p -> M=p+8*lh, N=lr) ----
  float bv[4];
#pragma unroll
  for (int ni = 0; ni < 4; ++ni) bv[ni] = bias[bn0 + ni * 16];

#pragma unroll
  for (int mi = 0; mi < 2; ++mi) {
#pragma unroll
    for (int ni = 0; ni < 4; ++ni) {
      int n = bn0 + ni * 16;
#pragma unroll
      for (int p = 0; p < 8; ++p) {
        int m = m0 + wm * 32 + mi * 16 + lh * 8 + p;
        if (m < n_edges) {
          float vvv = acc[mi][ni][p] + bv[ni];
          out[(size_t)m * NDIM + n] = fmaxf(vvv, 0.0f);
        }
      }
    }
  }
}

extern "C" void kernel_launch(void* const* d_in, const int* in_sizes, int n_in,
                              void* d_out, int out_size, void* d_ws, size_t ws_size,
                              hipStream_t stream) {
  const float* atom = (const float*)d_in[0];
  const float* edge = (const float*)d_in[1];
  const int*   src  = (const int*)d_in[2];
  const float* W    = (const float*)d_in[3];
  const float* bias = (const float*)d_in[4];
  float*       out  = (float*)d_out;

  const int n_edges = in_sizes[2];
  const size_t ws_needed = (size_t)2 * KDIM * NDIM * sizeof(unsigned short); // 98304 B
  const int use_ws = (ws_size >= ws_needed) ? 1 : 0;
  unsigned short* ws = (unsigned short*)d_ws;

  if (use_ws) wprep_kernel<<<1, 256, 0, stream>>>(W, ws);

  const int nblocks = (n_edges + MBLK - 1) / MBLK;
  const size_t lds_bytes = (size_t)4 * MBLK * SA * sizeof(unsigned short); // 204800 B
  a2b_kernel<<<nblocks, 256, lds_bytes, stream>>>(atom, edge, src, W, bias, ws,
                                                  out, n_edges, use_ws);
}
